// ModulatedConv2d_18262200943049
// MI455X (gfx1250) — compile-verified, matrix operand order checked by reading, not run on hardware
//
#include <hip/hip_runtime.h>
#include <hip/hip_bf16.h>
#include <math.h>

typedef __attribute__((ext_vector_type(16))) _Float16 v16h;
typedef __attribute__((ext_vector_type(8)))  float    v8f;
typedef __attribute__((ext_vector_type(4)))  int      v4i;

#define IN_CH   512
#define OUT_CH  512
#define HW      4096
#define KTOT    4608              // IN_CH * 9
#define BATCH   16
#define NSTEP   144               // KTOT / 32
#define CONV_SCALE (1.0f / 67.88225099390856f)   // 1/sqrt(512*9)
#define MOD_SCALE  (1.0f / 22.627416997969522f)  // 1/sqrt(512)

#if __has_builtin(__builtin_amdgcn_global_load_async_to_lds_b128)
#define USE_ASYNC_LDS 1
#define GLOBAL_AS(p) ((__attribute__((address_space(1))) v4i*)(void*)(p))
#define LDS_AS(p)    ((__attribute__((address_space(3))) v4i*)(void*)(p))
#else
#define USE_ASYNC_LDS 0
#endif

// ---------------------------------------------------------------------------
// 1) s[b][ic] = dot(style[b], mod_weight[ic]) * MOD_SCALE + mod_bias[ic]
// ---------------------------------------------------------------------------
__global__ __launch_bounds__(256) void style_mod_kernel(
    const float* __restrict__ style, const float* __restrict__ mod_weight,
    const float* __restrict__ mod_bias, float* __restrict__ s_out) {
  int t  = blockIdx.x * 256 + threadIdx.x;   // 0..8191
  int b  = t >> 9;
  int ic = t & 511;
  const float* st = style + b * 512;
  const float* mw = mod_weight + (size_t)ic * 512;
  float acc = 0.f;
  #pragma unroll 8
  for (int j = 0; j < 512; ++j) acc += st[j] * mw[j];
  s_out[t] = acc * MOD_SCALE + mod_bias[ic];
}

// ---------------------------------------------------------------------------
// 2) NCHW f32 -> NHWC f16 via LDS transpose (coalesced both sides)
// ---------------------------------------------------------------------------
__global__ __launch_bounds__(256) void nchw_to_nhwc_half(
    const float* __restrict__ in, _Float16* __restrict__ out) {
  __shared__ float tile[32 * 33];
  int b   = blockIdx.z;
  int ic0 = blockIdx.y * 32;
  int p0  = blockIdx.x * 32;
  int tx  = threadIdx.x & 31;
  int ty  = threadIdx.x >> 5;           // 0..7
  #pragma unroll
  for (int i = 0; i < 4; ++i) {
    int ic = ty + i * 8;
    tile[tx * 33 + ic] = in[(((size_t)b * IN_CH + ic0 + ic) << 12) + p0 + tx];
  }
  __syncthreads();
  #pragma unroll
  for (int i = 0; i < 4; ++i) {
    int pl = ty + i * 8;
    out[((size_t)((b << 12) + p0 + pl)) * IN_CH + ic0 + tx] =
        (_Float16)tile[pl * 33 + tx];
  }
}

// ---------------------------------------------------------------------------
// 3) Per (b, oc): modulate, demodulate, emit fp16 A-row  Ah[b][oc][tap*512+ic]
// ---------------------------------------------------------------------------
__global__ __launch_bounds__(256) void weight_prep(
    const float* __restrict__ weight, const float* __restrict__ s_buf,
    _Float16* __restrict__ Ah) {
  __shared__ float s_sh[512];
  __shared__ float red[256];
  int oc = blockIdx.x;
  int b  = blockIdx.y;
  int t  = threadIdx.x;
  s_sh[t]       = s_buf[b * 512 + t];
  s_sh[t + 256] = s_buf[b * 512 + t + 256];
  __syncthreads();

  const float* wrow = weight + (size_t)oc * KTOT;   // [ic][3][3] contiguous
  float sq = 0.f;
  for (int j = t; j < KTOT; j += 256) {
    int ic = j / 9;
    float wm = CONV_SCALE * wrow[j] * s_sh[ic];
    sq += wm * wm;
  }
  red[t] = sq;
  __syncthreads();
  for (int off = 128; off > 0; off >>= 1) {
    if (t < off) red[t] += red[t + off];
    __syncthreads();
  }
  float demod = rsqrtf(red[0] + 1e-8f);

  _Float16* arow = Ah + ((size_t)(b * OUT_CH + oc)) * KTOT;
  for (int kk = t; kk < KTOT; kk += 256) {
    int tap = kk >> 9;          // 0..8
    int ic  = kk & 511;
    float wm = CONV_SCALE * wrow[ic * 9 + tap] * s_sh[ic] * demod;
    arow[kk] = (_Float16)wm;
  }
}

// ---------------------------------------------------------------------------
// 4) Implicit GEMM, double-buffered LDS, one barrier per K-step.
//    Block tile 128x128, K-step 32; step s: tap = s>>4, A offset = s*32 halves,
//    B pointer maintained incrementally (+32 halves/step, +31264 leaving taps
//    2 and 5). 8 waves x (4x2) v_wmma_f32_16x16x32_f16.
//    Tile staging via GLOBAL_LOAD_ASYNC_TO_LDS (ASYNCcnt) when available.
// ---------------------------------------------------------------------------
__global__ __launch_bounds__(256) void mod_conv_gemm(
    const _Float16* __restrict__ Ah, const _Float16* __restrict__ Xh,
    float* __restrict__ out) {
  __shared__ _Float16 As[2][128 * 40];   // 2 x 10 KB
  __shared__ _Float16 Bs[2][128 * 40];   // 2 x 10 KB

  const int b       = blockIdx.z;
  const int oc_base = blockIdx.y * 128;
  const int n_base  = blockIdx.x * 128;
  const int t    = threadIdx.x;
  const int lane = t & 31;
  const int wave = t >> 5;
  const int wm   = wave & 1;          // 2 wave-rows (M)
  const int wn   = wave >> 1;         // 4 wave-cols (N)

  v8f acc[4][2];
  #pragma unroll
  for (int mt = 0; mt < 4; ++mt)
    #pragma unroll
    for (int nt = 0; nt < 2; ++nt) acc[mt][nt] = 0.f;

  // tile-load mapping: each thread owns one 16-half segment of one row/col
  const int row = t >> 1;             // 0..127
  const int seg = t & 1;              // 0/1: which 16-half segment of K=32
  const int ldsOff = row * 40 + seg * 16;
  const _Float16* aSrc =
      Ah + ((size_t)(b * OUT_CH + oc_base + row)) * KTOT + seg * 16;
  const int p = n_base + row;
  const int h = p >> 6, w = p & 63;
  // B pointer for the next step to fetch; starts at tap 0 => offset -65*512
  const _Float16* bptr =
      Xh + ((size_t)((b << 12) + p)) * IN_CH + seg * 16 - 65 * IN_CH;

  // per-thread tap validity mask (bit tap set if shifted pixel in-bounds)
  unsigned vmask = 0;
  #pragma unroll
  for (int tap = 0; tap < 9; ++tap) {
    int sh = h + tap / 3 - 1, sw = w + tap % 3 - 1;
    if (((unsigned)sh < 64u) && ((unsigned)sw < 64u)) vmask |= (1u << tap);
  }

  // fragment lane mapping (ISA 16-bit A/B layout)
  const int lml  = lane & 15;
  const int koff = (lane >> 4) * 8;

#if USE_ASYNC_LDS
  // -------- async-to-LDS staging: issue for step sn into buf --------
  auto issue = [&](int sn, int buf) {
    const _Float16* ap = aSrc + (sn << 5);
    _Float16* ad = &As[buf][ldsOff];
    __builtin_amdgcn_global_load_async_to_lds_b128(GLOBAL_AS(ap), LDS_AS(ad), 0, 0);
    __builtin_amdgcn_global_load_async_to_lds_b128(GLOBAL_AS(ap), LDS_AS(ad), 16, 0);
    _Float16* bd = &Bs[buf][ldsOff];
    if (vmask & (1u << (sn >> 4))) {
      __builtin_amdgcn_global_load_async_to_lds_b128(GLOBAL_AS(bptr), LDS_AS(bd), 0, 0);
      __builtin_amdgcn_global_load_async_to_lds_b128(GLOBAL_AS(bptr), LDS_AS(bd), 16, 0);
    } else {
      float4 z = make_float4(0.f, 0.f, 0.f, 0.f);
      ((float4*)bd)[0] = z;
      ((float4*)bd)[1] = z;
    }
    // advance B pointer to step sn+1
    bptr += (sn == 47 || sn == 95) ? 31264 : 32;
  };

  issue(0, 0);                         // prologue: stage step 0 into buffer 0

  for (int s = 0; s < NSTEP; ++s) {
    const int cur = s & 1;
    asm volatile("s_wait_asynccnt 0" ::: "memory");  // own async writes done
    __syncthreads();                   // buf[cur] complete; buf[cur^1] free

    if (s + 1 < NSTEP) issue(s + 1, cur ^ 1);

    const _Float16* Ac = As[cur];
    const _Float16* Bc = Bs[cur];
    v16h af[4], bf[2];
    #pragma unroll
    for (int mt = 0; mt < 4; ++mt) {
      const _Float16* src = Ac + (wm * 64 + mt * 16 + lml) * 40;
      ((float4*)&af[mt])[0] = *(const float4*)(src + koff);
      ((float4*)&af[mt])[1] = *(const float4*)(src + koff + 16);
    }
    #pragma unroll
    for (int nt = 0; nt < 2; ++nt) {
      const _Float16* src = Bc + (wn * 32 + nt * 16 + lml) * 40;
      ((float4*)&bf[nt])[0] = *(const float4*)(src + koff);
      ((float4*)&bf[nt])[1] = *(const float4*)(src + koff + 16);
    }
    #pragma unroll
    for (int mt = 0; mt < 4; ++mt)
      #pragma unroll
      for (int nt = 0; nt < 2; ++nt)
        acc[mt][nt] = __builtin_amdgcn_wmma_f32_16x16x32_f16(
            false, af[mt], false, bf[nt], (short)0, acc[mt][nt], false, false);
  }
#else
  // -------- register-staged fallback --------
  auto fetch = [&](int sn, float4& fa0, float4& fa1, float4& fb0, float4& fb1) {
    const float4* ap = (const float4*)(aSrc + (sn << 5));
    fa0 = ap[0];
    fa1 = ap[1];
    fb0 = make_float4(0.f, 0.f, 0.f, 0.f);
    fb1 = fb0;
    if (vmask & (1u << (sn >> 4))) {
      const float4* bp = (const float4*)bptr;
      fb0 = bp[0];
      fb1 = bp[1];
    }
    bptr += (sn == 47 || sn == 95) ? 31264 : 32;
  };
  auto stage = [&](int buf, const float4& fa0, const float4& fa1,
                   const float4& fb0, const float4& fb1) {
    float4* ad = (float4*)(&As[buf][ldsOff]);
    ad[0] = fa0;
    ad[1] = fa1;
    float4* bd = (float4*)(&Bs[buf][ldsOff]);
    bd[0] = fb0;
    bd[1] = fb1;
  };

  {
    float4 a0, a1, b0, b1;
    fetch(0, a0, a1, b0, b1);
    stage(0, a0, a1, b0, b1);
  }

  for (int s = 0; s < NSTEP; ++s) {
    const int cur = s & 1;
    __syncthreads();

    float4 na0, na1, nb0, nb1;
    const bool more = (s + 1) < NSTEP;
    if (more) fetch(s + 1, na0, na1, nb0, nb1);

    const _Float16* Ac = As[cur];
    const _Float16* Bc = Bs[cur];
    v16h af[4], bf[2];
    #pragma unroll
    for (int mt = 0; mt < 4; ++mt) {
      const _Float16* src = Ac + (wm * 64 + mt * 16 + lml) * 40;
      ((float4*)&af[mt])[0] = *(const float4*)(src + koff);
      ((float4*)&af[mt])[1] = *(const float4*)(src + koff + 16);
    }
    #pragma unroll
    for (int nt = 0; nt < 2; ++nt) {
      const _Float16* src = Bc + (wn * 32 + nt * 16 + lml) * 40;
      ((float4*)&bf[nt])[0] = *(const float4*)(src + koff);
      ((float4*)&bf[nt])[1] = *(const float4*)(src + koff + 16);
    }
    #pragma unroll
    for (int mt = 0; mt < 4; ++mt)
      #pragma unroll
      for (int nt = 0; nt < 2; ++nt)
        acc[mt][nt] = __builtin_amdgcn_wmma_f32_16x16x32_f16(
            false, af[mt], false, bf[nt], (short)0, acc[mt][nt], false, false);

    if (more) stage(cur ^ 1, na0, na1, nb0, nb1);
  }
#endif

  // ---- store C: VGPR r -> M = r + 8*(lane>=16), N = lane&15 ----
  const int m_half = (lane >> 4) * 8;
  #pragma unroll
  for (int mt = 0; mt < 4; ++mt)
    #pragma unroll
    for (int nt = 0; nt < 2; ++nt) {
      const int oc = oc_base + wm * 64 + mt * 16 + m_half;
      const int n  = n_base + wn * 32 + nt * 16 + lml;
      float* op = out + (((size_t)(b * OUT_CH + oc)) << 12) + n;
      #pragma unroll
      for (int r = 0; r < 8; ++r) op[(size_t)r << 12] = acc[mt][nt][r];
    }
}

// ---------------------------------------------------------------------------
extern "C" void kernel_launch(void* const* d_in, const int* in_sizes, int n_in,
                              void* d_out, int out_size, void* d_ws, size_t ws_size,
                              hipStream_t stream) {
  const float* input      = (const float*)d_in[0]; // (16,512,64,64)
  const float* style      = (const float*)d_in[1]; // (16,512)
  const float* weight     = (const float*)d_in[2]; // (1,512,512,3,3)
  const float* mod_weight = (const float*)d_in[3]; // (512,512)
  const float* mod_bias   = (const float*)d_in[4]; // (512,)
  float* out = (float*)d_out;

  char* ws = (char*)d_ws;
  const size_t A_bytes = (size_t)BATCH * OUT_CH * KTOT * sizeof(_Float16); // 75.5 MB
  const size_t X_bytes = (size_t)BATCH * HW * IN_CH * sizeof(_Float16);   // 67.1 MB
  _Float16* Ah   = (_Float16*)ws;
  _Float16* Xh   = (_Float16*)(ws + A_bytes);
  float*    sbuf = (float*)(ws + A_bytes + X_bytes);

  style_mod_kernel<<<32, 256, 0, stream>>>(style, mod_weight, mod_bias, sbuf);
  nchw_to_nhwc_half<<<dim3(128, 16, 16), 256, 0, stream>>>(input, Xh);
  weight_prep<<<dim3(512, 16), 256, 0, stream>>>(weight, sbuf, Ah);
  mod_conv_gemm<<<dim3(32, 4, 16), 256, 0, stream>>>(Ah, Xh, out);
}
